// MANN_62835371540516
// MI455X (gfx1250) — matrix operand order repaired and checked by reading, not compiled, not run on hardware
//
#include <hip/hip_runtime.h>
#include <hip/hip_bf16.h>
#include <stdint.h>

// ---------------------------------------------------------------------------
// Problem dims
// ---------------------------------------------------------------------------
#define B_      2048
#define IN_DIM  512
#define CTRL    512
#define N_LOC   65536
#define LOC_SZ  64

typedef __attribute__((ext_vector_type(16))) __bf16 v16bf;
typedef __attribute__((ext_vector_type(8)))  float  v8f;

union Frag16 {
    v16bf v;
    unsigned short u[16];
};

__device__ __forceinline__ unsigned short f2bf(float f) {
    union { float f; uint32_t u; } c; c.f = f;
    uint32_t u = c.u;
    u += 0x7fffu + ((u >> 16) & 1u);      // round-to-nearest-even
    return (unsigned short)(u >> 16);
}
__device__ __forceinline__ float bf2f(unsigned short h) {
    union { uint32_t u; float f; } c; c.u = ((uint32_t)h) << 16;
    return c.f;
}

__device__ __forceinline__ v8f wmma_bf16(v16bf a, v16bf b, v8f c) {
    // D = A(16x32 bf16) * B(32x16 bf16) + C(16x16 f32)
    return __builtin_amdgcn_wmma_f32_16x16x32_bf16(
        false, a, false, b, (short)0, c, false, false);
}

// Constant all-ones B fragment (bf16 1.0 = 0x3F80): D = A @ ones -> row sums.
__device__ __forceinline__ v16bf ones_frag() {
    Frag16 f;
#pragma unroll
    for (int i = 0; i < 16; ++i) f.u[i] = 0x3F80;
    return f.v;
}

// A fragment: 16x32 bf16, source row-major [M x K], leading dim lda.
// lane m = lane%16, half = lane/16; reg v: K = (v&4?16:0) + half*8 + (v&3)*2 +{0,1}
// -> two contiguous 8-ushort runs per lane -> b128 loads.
__device__ __forceinline__ v16bf load_a(const unsigned short* p, int arow,
                                        int lda, int kbase, int lane) {
    Frag16 f;
    const int hl = lane >> 4;
    const unsigned short* rp = p + (size_t)arow * lda + kbase;
#pragma unroll
    for (int v = 0; v < 8; ++v) {
        int k = ((v & 4) ? 16 : 0) + hl * 8 + (v & 3) * 2;
        f.u[2 * v]     = rp[k];
        f.u[2 * v + 1] = rp[k + 1];
    }
    return f.v;
}

// B fragment: 32x16 bf16 (K x N) from an N-major source (row-major [N x K]).
// lane n = nrow0 + lane%16, half = lane/16; reg v: K = (v&4?16:0) + half*8 + (v&3)*2 +{0,1}
// -> two contiguous 8-ushort runs per lane -> b128 loads.
__device__ __forceinline__ v16bf load_bt(const unsigned short* p, int nrow0,
                                         int ldk, int kbase, int lane) {
    Frag16 f;
    const int hl = lane >> 4;
    const unsigned short* rp = p + (size_t)(nrow0 + (lane & 15)) * ldk + kbase;
#pragma unroll
    for (int v = 0; v < 8; ++v) {
        int k = ((v & 4) ? 16 : 0) + hl * 8 + (v & 3) * 2;
        f.u[2 * v]     = rp[k];
        f.u[2 * v + 1] = rp[k + 1];
    }
    return f.v;
}

// ---------------------------------------------------------------------------
// Prep kernels
// ---------------------------------------------------------------------------
__global__ void cast_bf16_kernel(const float* __restrict__ src,
                                 unsigned short* __restrict__ dst, int n) {
    int i = blockIdx.x * blockDim.x + threadIdx.x;
    if (i < n) dst[i] = f2bf(src[i]);
}

// Tiled transpose + cast: src f32 [K][N] -> dst bf16 [N][K]
__global__ void transpose_cast_kernel(const float* __restrict__ src,
                                      unsigned short* __restrict__ dst,
                                      int K, int N) {
    __shared__ unsigned short tile[32][33];
    int kb = blockIdx.y * 32;
    int nb = blockIdx.x * 32;
    int tx = threadIdx.x & 31;
    int ty = threadIdx.x >> 5;    // 0..7
#pragma unroll
    for (int i = 0; i < 32; i += 8)
        tile[ty + i][tx] = f2bf(src[(size_t)(kb + ty + i) * N + nb + tx]);
    __syncthreads();
#pragma unroll
    for (int i = 0; i < 32; i += 8)
        dst[(size_t)(nb + ty + i) * K + kb + tx] = tile[tx][ty + i];
}

// Per-row norm of memory: emit Mn_bf16 (normalized rows, row-major)
__global__ void prepM_kernel(const float* __restrict__ M,
                             unsigned short* __restrict__ Mnb) {
    int wave = (blockIdx.x * blockDim.x + threadIdx.x) >> 5;
    int lane = threadIdx.x & 31;
    if (wave >= N_LOC) return;
    const float* row = M + (size_t)wave * LOC_SZ;
    float a = row[lane];
    float b = row[lane + 32];
    float s = a * a + b * b;
#pragma unroll
    for (int m = 1; m < 32; m <<= 1) s += __shfl_xor(s, m, 32);
    float inv = 1.0f / (sqrtf(s) + 1e-8f);
    size_t base = (size_t)wave * LOC_SZ;
    Mnb[base + lane]      = f2bf(a * inv);
    Mnb[base + lane + 32] = f2bf(b * inv);
}

// ---------------------------------------------------------------------------
// Controller: h = tanh(x @ W_h + b_h)   [2048,512]x[512,512]
// ---------------------------------------------------------------------------
__global__ void ctrl_gemm_kernel(const unsigned short* __restrict__ xb,
                                 const unsigned short* __restrict__ Whbt,
                                 const float* __restrict__ b_h,
                                 float* __restrict__ h_f32,
                                 unsigned short* __restrict__ h_bf16) {
    int wave = (blockIdx.x * blockDim.x + threadIdx.x) >> 5;
    int lane = threadIdx.x & 31;
    int rt = wave >> 5;        // 0..127 row tile
    int ct = wave & 31;        // 0..31 col tile
    int row0 = rt * 16, col0 = ct * 16;

    v8f c = {};
    for (int ks = 0; ks < IN_DIM; ks += 32) {
        v16bf a = load_a(xb, row0 + (lane & 15), IN_DIM, ks, lane);
        v16bf b = load_bt(Whbt, col0, IN_DIM, ks, lane);
        c = wmma_bf16(a, b, c);
    }
    int n = col0 + (lane & 15);
    int hl = lane >> 4;
    float bias = b_h[n];
#pragma unroll
    for (int v = 0; v < 8; ++v) {
        int m = row0 + hl * 8 + v;
        float val = tanhf(c[v] + bias);
        h_f32[(size_t)m * CTRL + n]  = val;
        h_bf16[(size_t)m * CTRL + n] = f2bf(val);
    }
}

// ---------------------------------------------------------------------------
// Read key: key = tanh(h @ W_k + b_k), kn = key / (||key|| + eps)
// ---------------------------------------------------------------------------
__global__ void key_kernel(const unsigned short* __restrict__ hb,
                           const unsigned short* __restrict__ Wkbt,
                           const float* __restrict__ b_k,
                           unsigned short* __restrict__ kn_bf16) {
    __shared__ float rowsq[16];
    int lane = threadIdx.x & 31;
    int w = threadIdx.x >> 5;          // 0..3 -> N tile
    int row0 = blockIdx.x * 16;
    int col0 = w * 16;

    v8f c = {};
    for (int ks = 0; ks < CTRL; ks += 32) {
        v16bf a = load_a(hb, row0 + (lane & 15), CTRL, ks, lane);
        v16bf b = load_bt(Wkbt, col0, CTRL, ks, lane);
        c = wmma_bf16(a, b, c);
    }
    int n = col0 + (lane & 15);
    int hl = lane >> 4;
    float bias = b_k[n];
    float key[8], sq[8];
#pragma unroll
    for (int v = 0; v < 8; ++v) {
        key[v] = tanhf(c[v] + bias);
        float s = key[v] * key[v];
        s += __shfl_xor(s, 1, 32);
        s += __shfl_xor(s, 2, 32);
        s += __shfl_xor(s, 4, 32);
        s += __shfl_xor(s, 8, 32);
        sq[v] = s;
    }
    if (threadIdx.x < 16) rowsq[threadIdx.x] = 0.f;
    __syncthreads();
    if ((lane & 15) == 0) {
#pragma unroll
        for (int v = 0; v < 8; ++v) atomicAdd(&rowsq[hl * 8 + v], sq[v]);
    }
    __syncthreads();
#pragma unroll
    for (int v = 0; v < 8; ++v) {
        int m = row0 + hl * 8 + v;
        float norm = sqrtf(rowsq[hl * 8 + v]) + 1e-8f;
        kn_bf16[(size_t)m * LOC_SZ + n] = f2bf(key[v] / norm);
    }
}

// ---------------------------------------------------------------------------
// Fused attention over memory (sim in [-1,1] => fixed softmax shift of 1.0,
// no online-max, row sums via an all-ones WMMA tile):
//   p = exp(kn@Mn^T - 1) ; r = (p @ M) / (p @ 1)
// block = 128 threads = 4 waves, 32 queries (2 row tiles) per block.
// Each wave owns 32 locations per iteration; B frags shared by both row tiles.
// ---------------------------------------------------------------------------
__global__ void flash_kernel(const unsigned short* __restrict__ knb,
                             const unsigned short* __restrict__ Mnb,
                             const unsigned short* __restrict__ Mtb,
                             float* __restrict__ r_out,
                             float* __restrict__ rowmax_out,
                             float* __restrict__ rowsum_out) {
    __shared__ unsigned short pbuf[4 * 2 * 16 * 32];   // [wave][tile][16][32]
    __shared__ float lsum[4][32];
    __shared__ float lr[4][32 * LOC_SZ];

    int lane = threadIdx.x & 31;
    int w = threadIdx.x >> 5;
    int hl = lane >> 4;
    int qrow0 = blockIdx.x * 32;

    // query A fragments for two 16-row tiles (K = 64 -> two frags each)
    v16bf aq[2][2];
#pragma unroll
    for (int t = 0; t < 2; ++t) {
        aq[t][0] = load_a(knb, qrow0 + t * 16 + (lane & 15), LOC_SZ, 0, lane);
        aq[t][1] = load_a(knb, qrow0 + t * 16 + (lane & 15), LOC_SZ, 32, lane);
    }
    const v16bf bones = ones_frag();

    // accumulators: per row tile, 4 value tiles + 1 row-sum tile
    v8f racc[2][5];
#pragma unroll
    for (int t = 0; t < 2; ++t)
#pragma unroll
        for (int j = 0; j < 5; ++j) racc[t][j] = (v8f){};

    const int iters = N_LOC / 128;
    for (int it = 0; it < iters; ++it) {
        int base = it * 128 + w * 32;

        // prefetch next chunk (global_prefetch_b8)
        if (it + 1 < iters) {
            int nb2 = base + 128;
            __builtin_prefetch(Mnb + (size_t)(nb2 + lane) * LOC_SZ, 0, 1);
            __builtin_prefetch(Mtb + (size_t)lane * N_LOC + nb2, 0, 1);
            __builtin_prefetch(Mtb + (size_t)(lane + 32) * N_LOC + nb2, 0, 1);
        }

        // ---- GEMM1 B fragments (shared by both row tiles) ----
        v16bf bn00 = load_bt(Mnb, base,      LOC_SZ, 0,  lane);
        v16bf bn01 = load_bt(Mnb, base,      LOC_SZ, 32, lane);
        v16bf bn10 = load_bt(Mnb, base + 16, LOC_SZ, 0,  lane);
        v16bf bn11 = load_bt(Mnb, base + 16, LOC_SZ, 32, lane);

#pragma unroll
        for (int t = 0; t < 2; ++t) {
            v8f c0 = {}, c1 = {};
            c0 = wmma_bf16(aq[t][0], bn00, c0);
            c0 = wmma_bf16(aq[t][1], bn01, c0);
            c1 = wmma_bf16(aq[t][0], bn10, c1);
            c1 = wmma_bf16(aq[t][1], bn11, c1);
            // p = exp(sim - 1): sim bounded by 1, softmax shift-invariant
            unsigned short* pb = pbuf + (w * 2 + t) * 512;
#pragma unroll
            for (int v = 0; v < 8; ++v) {
                int m = hl * 8 + v;
                pb[m * 32 + (lane & 15)]      = f2bf(__expf(c0[v] - 1.0f));
                pb[m * 32 + 16 + (lane & 15)] = f2bf(__expf(c1[v] - 1.0f));
            }
        }
        __syncthreads();

        // ---- GEMM2: r += P @ M_chunk, rowsum += P @ 1 ----
        v16bf pa0 = load_a(pbuf + (w * 2 + 0) * 512, (lane & 15), 32, 0, lane);
        v16bf pa1 = load_a(pbuf + (w * 2 + 1) * 512, (lane & 15), 32, 0, lane);
#pragma unroll
        for (int j = 0; j < 4; ++j) {
            v16bf bm = load_bt(Mtb, j * 16, N_LOC, base, lane);
            racc[0][j] = wmma_bf16(pa0, bm, racc[0][j]);
            racc[1][j] = wmma_bf16(pa1, bm, racc[1][j]);
        }
        racc[0][4] = wmma_bf16(pa0, bones, racc[0][4]);
        racc[1][4] = wmma_bf16(pa1, bones, racc[1][4]);
        __syncthreads();     // protect pbuf WAR for next iteration
    }

    // ---- merge 4 waves ----
#pragma unroll
    for (int t = 0; t < 2; ++t) {
        if ((lane & 15) == 0) {
#pragma unroll
            for (int v = 0; v < 8; ++v)
                lsum[w][t * 16 + hl * 8 + v] = racc[t][4][v];
        }
#pragma unroll
        for (int j = 0; j < 4; ++j) {
            int n = j * 16 + (lane & 15);
#pragma unroll
            for (int v = 0; v < 8; ++v)
                lr[w][(t * 16 + hl * 8 + v) * LOC_SZ + n] = racc[t][j][v];
        }
    }
    __syncthreads();

    int tid = threadIdx.x;
    int m = tid >> 2;                 // 0..31
    int c0i = (tid & 3) * 16;         // 16 cols per thread
    float gsum = 0.f;
#pragma unroll
    for (int ww = 0; ww < 4; ++ww) gsum += lsum[ww][m];
    float inv = 1.0f / gsum;
#pragma unroll
    for (int cc = 0; cc < 16; ++cc) {
        float acc = 0.f;
#pragma unroll
        for (int ww = 0; ww < 4; ++ww)
            acc += lr[ww][m * LOC_SZ + c0i + cc];
        r_out[(size_t)(qrow0 + m) * LOC_SZ + c0i + cc] = acc * inv;
    }
    if ((tid & 3) == 0) {
        rowmax_out[qrow0 + m] = 1.0f;     // fixed softmax shift
        rowsum_out[qrow0 + m] = gsum;
    }
}

// ---------------------------------------------------------------------------
// output[b] = h[b,:] . W_out[:512] + r[b,:] . W_out[512:576] + b_out
// ---------------------------------------------------------------------------
__global__ void out_kernel(const float* __restrict__ h,
                           const float* __restrict__ r,
                           const float* __restrict__ W_out,
                           const float* __restrict__ b_out,
                           float* __restrict__ out) {
    int wave = (blockIdx.x * blockDim.x + threadIdx.x) >> 5;
    int lane = threadIdx.x & 31;
    if (wave >= B_) return;
    float acc = 0.f;
    for (int j = lane; j < CTRL; j += 32)
        acc += h[(size_t)wave * CTRL + j] * W_out[j];
    for (int j = lane; j < LOC_SZ; j += 32)
        acc += r[(size_t)wave * LOC_SZ + j] * W_out[CTRL + j];
#pragma unroll
    for (int m = 1; m < 32; m <<= 1) acc += __shfl_xor(acc, m, 32);
    if (lane == 0) out[wave] = acc + b_out[0];
}

// ---------------------------------------------------------------------------
// extras: h[-1] (512 floats), gate[-1] = x[-1] . W_g + b_g
// ---------------------------------------------------------------------------
__global__ void extras_kernel(const float* __restrict__ x,
                              const float* __restrict__ W_g,
                              const float* __restrict__ b_g,
                              const float* __restrict__ h_f32,
                              float* __restrict__ out_hlast,
                              float* __restrict__ out_gate) {
    int tid = threadIdx.x;   // 512 threads
    out_hlast[tid] = h_f32[(size_t)(B_ - 1) * CTRL + tid];
    if (tid < 32) {
        float acc = 0.f;
        for (int j = tid; j < IN_DIM; j += 32)
            acc += x[(size_t)(B_ - 1) * IN_DIM + j] * W_g[j];
#pragma unroll
        for (int m = 1; m < 32; m <<= 1) acc += __shfl_xor(acc, m, 32);
        if (tid == 0) out_gate[0] = acc + b_g[0];
    }
}

// w_read[-1][l] = exp(sim[-1,l] - shift) / sum
__global__ void wlast_kernel(const unsigned short* __restrict__ knb,
                             const unsigned short* __restrict__ Mnb,
                             const float* __restrict__ rowmax,
                             const float* __restrict__ rowsum,
                             float* __restrict__ wout) {
    int l = blockIdx.x * blockDim.x + threadIdx.x;
    if (l >= N_LOC) return;
    float acc = 0.f;
#pragma unroll 8
    for (int d = 0; d < LOC_SZ; ++d)
        acc += bf2f(knb[(size_t)(B_ - 1) * LOC_SZ + d]) *
               bf2f(Mnb[(size_t)l * LOC_SZ + d]);
    wout[l] = __expf(acc - rowmax[B_ - 1]) / rowsum[B_ - 1];
}

// ---------------------------------------------------------------------------
// launch
// ---------------------------------------------------------------------------
extern "C" void kernel_launch(void* const* d_in, const int* in_sizes, int n_in,
                              void* d_out, int out_size, void* d_ws, size_t ws_size,
                              hipStream_t stream) {
    const float* x     = (const float*)d_in[0];
    const float* W_h   = (const float*)d_in[1];
    const float* b_h   = (const float*)d_in[2];
    const float* W_g   = (const float*)d_in[3];
    const float* b_g   = (const float*)d_in[4];
    const float* W_k   = (const float*)d_in[5];
    const float* b_k   = (const float*)d_in[6];
    const float* M     = (const float*)d_in[7];
    const float* W_out = (const float*)d_in[8];
    const float* b_out = (const float*)d_in[9];

    float* out = (float*)d_out;
    float* out_hlast = out + B_;              // 512
    float* out_gate  = out + B_ + CTRL;       // 1
    float* out_wlast = out + B_ + CTRL + 1;   // 65536

    char* ws = (char*)d_ws;
    size_t off = 0;
    auto alloc = [&](size_t bytes) -> char* {
        char* p = ws + off;
        off += (bytes + 255) & ~(size_t)255;
        return p;
    };
    unsigned short* xb   = (unsigned short*)alloc((size_t)B_ * IN_DIM * 2);
    unsigned short* Whbt = (unsigned short*)alloc((size_t)IN_DIM * CTRL * 2);  // [CTRL][IN_DIM]
    unsigned short* Wkbt = (unsigned short*)alloc((size_t)CTRL * LOC_SZ * 2);  // [LOC_SZ][CTRL]
    unsigned short* Mnb  = (unsigned short*)alloc((size_t)N_LOC * LOC_SZ * 2); // [loc][64]
    unsigned short* Mtb  = (unsigned short*)alloc((size_t)N_LOC * LOC_SZ * 2); // [64][loc]
    float*          h32  = (float*)alloc((size_t)B_ * CTRL * 4);
    unsigned short* hb   = (unsigned short*)alloc((size_t)B_ * CTRL * 2);
    unsigned short* knb  = (unsigned short*)alloc((size_t)B_ * LOC_SZ * 2);
    float*          r32  = (float*)alloc((size_t)B_ * LOC_SZ * 4);
    float*          rmax = (float*)alloc((size_t)B_ * 4);
    float*          rsum = (float*)alloc((size_t)B_ * 4);
    (void)ws_size; (void)in_sizes; (void)n_in; (void)out_size;

    // prep: casts + transposes
    cast_bf16_kernel<<<(B_ * IN_DIM + 255) / 256, 256, 0, stream>>>(x, xb, B_ * IN_DIM);
    {   // W_h [512][512] f32 -> Whbt [512][512] bf16 (transposed)
        dim3 g(CTRL / 32, IN_DIM / 32);
        transpose_cast_kernel<<<g, 256, 0, stream>>>(W_h, Whbt, IN_DIM, CTRL);
    }
    {   // W_k [512][64] f32 -> Wkbt [64][512] bf16 (transposed)
        dim3 g(LOC_SZ / 32, CTRL / 32);
        transpose_cast_kernel<<<g, 256, 0, stream>>>(W_k, Wkbt, CTRL, LOC_SZ);
    }
    {   // M [65536][64] f32 -> Mtb [64][65536] bf16 (transposed, raw values)
        dim3 g(LOC_SZ / 32, N_LOC / 32);
        transpose_cast_kernel<<<g, 256, 0, stream>>>(M, Mtb, N_LOC, LOC_SZ);
    }
    prepM_kernel<<<(N_LOC * 32) / 256, 256, 0, stream>>>(M, Mnb);

    // controller GEMM: 128x32 tiles -> 4096 waves -> 512 blocks x 256
    ctrl_gemm_kernel<<<512, 256, 0, stream>>>(xb, Whbt, b_h, h32, hb);

    // key GEMM + normalize: 128 blocks x 128
    key_kernel<<<B_ / 16, 128, 0, stream>>>(hb, Wkbt, b_k, knb);

    // fused attention over memory: 32 queries/block -> 64 blocks x 128
    flash_kernel<<<B_ / 32, 128, 0, stream>>>(knb, Mnb, Mtb, r32, rmax, rsum);

    // final projection: 2048 waves -> 256 blocks x 256
    out_kernel<<<(B_ * 32) / 256, 256, 0, stream>>>(h32, r32, W_out, b_out, out);

    // extras
    extras_kernel<<<1, 512, 0, stream>>>(x, W_g, b_g, h32, out_hlast, out_gate);
    wlast_kernel<<<N_LOC / 256, 256, 0, stream>>>(knb, Mnb, rmax, rsum, out_wlast);
}
